// StackedGRUModel_33182917329284
// MI455X (gfx1250) — compile-verified, matrix operand order checked by reading, not run on hardware
//
#include <hip/hip_runtime.h>
#include <stdint.h>

// ---------------------------------------------------------------------------
// Fused 3-layer GRU stack for MI455X (gfx1250, wave32, WMMA bf16 16x16x32).
// One workgroup (4 waves) owns a 16-row batch tile for all 512 time steps.
// All weight fragments live in VGPRs across the t-loop (occupancy is capped
// at 1 wave/SIMD by the B/16=64 chain count, so registers are free).
// ---------------------------------------------------------------------------

typedef __attribute__((ext_vector_type(16))) __bf16 v16bf;
typedef __attribute__((ext_vector_type(2)))  __bf16 v2bf;
typedef __attribute__((ext_vector_type(8)))  float  v8f;

union FragU {
  v16bf    v;
  uint32_t u[8];
};

static constexpr int GRU_T  = 512;
static constexpr int GRU_D0 = 32;
static constexpr int GRU_H0 = 64;
static constexpr int GRU_H1 = 32;
static constexpr int GRU_H2 = 16;

// Native f32->bf16 conversions (v_cvt_pk_bf16_f32 on gfx1250).
__device__ __forceinline__ uint32_t pack_bf16(float lo, float hi) {
  v2bf p;
  p[0] = (__bf16)lo;
  p[1] = (__bf16)hi;
  return __builtin_bit_cast(uint32_t, p);
}
__device__ __forceinline__ uint16_t bf16_bits(float f) {
  __bf16 b = (__bf16)f;
  return __builtin_bit_cast(uint16_t, b);
}

// K index of the low 16 bits of fragment VGPR v for this lane half
// (CDNA5 ISA 7.12.2, 16-bit A/B layout). High half is +1.
__device__ __forceinline__ int kloc0(int v, int hi) {
  int base = hi ? 8 : 0;
  return (v < 4) ? (base + 2 * v) : (base + 16 + 2 * (v - 4));
}

__device__ __forceinline__ v8f wmma_bf16(v16bf a, v16bf b, v8f c) {
  return __builtin_amdgcn_wmma_f32_16x16x32_bf16(
      /*neg_a=*/false, a, /*neg_b=*/false, b,
      /*c_mod=*/(short)0, c, /*reuse_a=*/false, /*reuse_b=*/false);
}

// Fast activations: v_exp_f32 + v_rcp_f32, no precise-division fixup sequence.
__device__ __forceinline__ float sigmoidf_fast(float x) {
  return __builtin_amdgcn_rcpf(1.0f + __expf(-x));
}
__device__ __forceinline__ float tanhf_fast(float x) {
  return 2.0f * __builtin_amdgcn_rcpf(1.0f + __expf(-2.0f * x)) - 1.0f;
}

// Build one 32x16 (KxN) bf16 B-fragment from a row-major [Kdim x Ncols] f32
// weight matrix. Rows beyond Kdim are zero (used to pad K=16 up to 32).
__device__ __forceinline__ FragU buildFrag(const float* __restrict__ P, int Kdim,
                                           int Ncols, int n0, int kt, int lane) {
  FragU f;
  int hi = lane >> 4;
  int n  = n0 + (lane & 15);
#pragma unroll
  for (int v = 0; v < 8; ++v) {
    int k = kt * 32 + kloc0(v, hi);
    float lo = (k < Kdim)     ? P[(size_t)k * Ncols + n]       : 0.0f;
    float hw = (k + 1 < Kdim) ? P[(size_t)(k + 1) * Ncols + n] : 0.0f;
    f.u[v] = pack_bf16(lo, hw);
  }
  return f;
}

// Gather a 16x32 (MxK) bf16 A-fragment from a padded bf16 state tile in LDS.
template <int S>
__device__ __forceinline__ v16bf gatherAh(const uint16_t (&h)[16][S], int Kdim,
                                          int kt, int lane) {
  FragU f;
  int M = lane & 15, hi = lane >> 4;
#pragma unroll
  for (int v = 0; v < 8; ++v) {
    int k = kt * 32 + kloc0(v, hi);
    uint32_t pk = 0;
    if (k < Kdim) pk = *(const uint32_t*)&h[M][k];   // consecutive K pair
    f.u[v] = pk;
  }
  return f.v;
}

// Gather A-fragment for the layer-0 input x_t (fp32 in LDS -> bf16 pairs).
template <int S>
__device__ __forceinline__ v16bf gatherAx(const float (&xs)[16][S], int lane) {
  FragU f;
  int M = lane & 15, hi = lane >> 4;
#pragma unroll
  for (int v = 0; v < 8; ++v) {
    int k = kloc0(v, hi);
    f.u[v] = pack_bf16(xs[M][k], xs[M][k + 1]);
  }
  return f.v;
}

__device__ __forceinline__ v16bf loadFrag(const uint4 (&fr)[54][32][2], int id,
                                          int lane) {
  FragU f;
  ((uint4*)f.u)[0] = fr[id][lane][0];   // 2x ds_load_b128
  ((uint4*)f.u)[1] = fr[id][lane][1];
  return f.v;
}

__global__ __launch_bounds__(128)
void stacked_gru_fused(const float* __restrict__ x,
                       const float* __restrict__ W0, const float* __restrict__ U0,
                       const float* __restrict__ bi0, const float* __restrict__ br0,
                       const float* __restrict__ W1, const float* __restrict__ U1,
                       const float* __restrict__ bi1, const float* __restrict__ br1,
                       const float* __restrict__ W2, const float* __restrict__ U2,
                       const float* __restrict__ bi2, const float* __restrict__ br2,
                       const float* __restrict__ Wd, const float* __restrict__ bd,
                       float* __restrict__ out) {
  // Weight fragments: W0(12) U0(24) W1(12) U1(6) = 54 frags, 1KB each.
  __shared__ uint4    fragLDS[54][32][2];   // 55296 B (staging for reg-resident frags)
  __shared__ float    xs[16][36];           // x_t stage (padded, 16B-aligned rows)
  __shared__ uint16_t h0bf[16][68];         // bf16 hidden states (padded rows)
  __shared__ uint16_t h1bf[16][36];
  __shared__ uint16_t h2bf[16][20];
  __shared__ float    h2f[16][17];          // final h2 for dense head

  const int tid  = threadIdx.x;
  const int lane = tid & 31;
  const int wave = __builtin_amdgcn_readfirstlane(tid >> 5);  // scalar branchy
  const int c    = lane & 15;
  const int hi   = lane >> 4;
  const int b0   = blockIdx.x * 16;

  // ---- Stage 0: pre-swizzle weights into WMMA B-fragment layout ----
  for (int id = wave; id < 54; id += 4) {
    FragU f;
    if (id < 12)      { int g = id / 4, ct = id % 4;
                        f = buildFrag(W0, GRU_D0, 3 * GRU_H0, g * GRU_H0 + ct * 16, 0, lane); }
    else if (id < 36) { int r = id - 12; int kt = r & 1; r >>= 1; int g = r / 4, ct = r % 4;
                        f = buildFrag(U0, GRU_H0, 3 * GRU_H0, g * GRU_H0 + ct * 16, kt, lane); }
    else if (id < 48) { int r = id - 36; int kt = r & 1; r >>= 1; int g = r / 2, ct = r % 2;
                        f = buildFrag(W1, GRU_H0, 3 * GRU_H1, g * GRU_H1 + ct * 16, kt, lane); }
    else              { int r = id - 48; int g = r / 2, ct = r % 2;
                        f = buildFrag(U1, GRU_H1, 3 * GRU_H1, g * GRU_H1 + ct * 16, 0, lane); }
    fragLDS[id][lane][0] = ((const uint4*)f.u)[0];
    fragLDS[id][lane][1] = ((const uint4*)f.u)[1];
  }
  // Layer-2 weights: build fragments straight into VGPRs.
  v16bf w2f[3], u2f[3];
#pragma unroll
  for (int g = 0; g < 3; ++g) {
    w2f[g] = buildFrag(W2, GRU_H1, 3 * GRU_H2, g * GRU_H2, 0, lane).v;
    u2f[g] = buildFrag(U2, GRU_H2, 3 * GRU_H2, g * GRU_H2, 0, lane).v;  // K=16 zero-padded
  }

  // Zero hidden-state staging (h_0 = 0).
  for (int i = tid; i < 16 * 68; i += 128) (&h0bf[0][0])[i] = 0;
  for (int i = tid; i < 16 * 36; i += 128) (&h1bf[0][0])[i] = 0;
  for (int i = tid; i < 16 * 20; i += 128) (&h2bf[0][0])[i] = 0;

  // Per-lane gate biases (wave w owns gate columns [16w,16w+16) of each gate).
  const int j0 = wave * 16 + c;
  const float bz0  = bi0[j0] + br0[j0];
  const float brr0 = bi0[GRU_H0 + j0] + br0[GRU_H0 + j0];
  const float bxh0 = bi0[2 * GRU_H0 + j0];
  const float bhh0 = br0[2 * GRU_H0 + j0];
  const int ct1 = (wave < 2) ? wave : 0;           // clamp: waves 2,3 stay in-bounds
  const int j1  = ct1 * 16 + c;
  const float bz1  = bi1[j1] + br1[j1];
  const float brr1 = bi1[GRU_H1 + j1] + br1[GRU_H1 + j1];
  const float bxh1 = bi1[2 * GRU_H1 + j1];
  const float bhh1 = br1[2 * GRU_H1 + j1];
  const float bz2  = bi2[c] + br2[c];
  const float brr2 = bi2[GRU_H2 + c] + br2[GRU_H2 + c];
  const float bxh2 = bi2[2 * GRU_H2 + c];
  const float bhh2 = br2[2 * GRU_H2 + c];

  // fp32 recurrent state lives in D-fragment registers of the owning wave.
  float h0r[8], h1r[8], h2r[8];
#pragma unroll
  for (int i = 0; i < 8; ++i) { h0r[i] = 0.f; h1r[i] = 0.f; h2r[i] = 0.f; }

  // x prefetch pipeline: each thread owns a float4 of the 16x32 tile.
  const int xr = tid >> 3, xc = (tid & 7) * 4;
  const float* xbase = x + (size_t)(b0 + xr) * GRU_T * GRU_D0 + xc;
  *(float4*)&xs[xr][xc] = *(const float4*)(xbase);        // x_0 -> LDS
  float4 pf = *(const float4*)(xbase + GRU_D0);           // x_1 -> regs
  __syncthreads();

  // ---- Hoist ALL weight fragments into registers (loop-invariant) ----------
  // f0[0..2] = W0 z/r/h ; f0[3..8] = U0 z0,z1,r0,r1,h0,h1   (gate tile = wave)
  // f1[0..2] = U1 z/r/h ; f1[3..8] = W1 z0,z1,r0,r1,h0,h1   (gate tile = ct1)
  v16bf f0[9], f1[9];
  {
    const int ct = wave;
    f0[0] = loadFrag(fragLDS, 0 + ct, lane);
    f0[1] = loadFrag(fragLDS, 4 + ct, lane);
    f0[2] = loadFrag(fragLDS, 8 + ct, lane);
    f0[3] = loadFrag(fragLDS, 12 + (0 * 4 + ct) * 2 + 0, lane);
    f0[4] = loadFrag(fragLDS, 12 + (0 * 4 + ct) * 2 + 1, lane);
    f0[5] = loadFrag(fragLDS, 12 + (1 * 4 + ct) * 2 + 0, lane);
    f0[6] = loadFrag(fragLDS, 12 + (1 * 4 + ct) * 2 + 1, lane);
    f0[7] = loadFrag(fragLDS, 12 + (2 * 4 + ct) * 2 + 0, lane);
    f0[8] = loadFrag(fragLDS, 12 + (2 * 4 + ct) * 2 + 1, lane);
    f1[0] = loadFrag(fragLDS, 48 + 0 * 2 + ct1, lane);
    f1[1] = loadFrag(fragLDS, 48 + 1 * 2 + ct1, lane);
    f1[2] = loadFrag(fragLDS, 48 + 2 * 2 + ct1, lane);
    f1[3] = loadFrag(fragLDS, 36 + (0 * 2 + ct1) * 2 + 0, lane);
    f1[4] = loadFrag(fragLDS, 36 + (0 * 2 + ct1) * 2 + 1, lane);
    f1[5] = loadFrag(fragLDS, 36 + (1 * 2 + ct1) * 2 + 0, lane);
    f1[6] = loadFrag(fragLDS, 36 + (1 * 2 + ct1) * 2 + 1, lane);
    f1[7] = loadFrag(fragLDS, 36 + (2 * 2 + ct1) * 2 + 0, lane);
    f1[8] = loadFrag(fragLDS, 36 + (2 * 2 + ct1) * 2 + 1, lane);
  }

  for (int t = 0; t < GRU_T; ++t) {
    // ---- Phase A: gather A-fragments for step t (state h_{t-1}, input x_t) --
    v16bf ax   = gatherAx(xs, lane);
    v16bf ah0a = gatherAh(h0bf, GRU_H0, 0, lane);
    v16bf ah0b = gatherAh(h0bf, GRU_H0, 1, lane);
    v16bf ah1  = gatherAh(h1bf, GRU_H1, 0, lane);
    __syncthreads();                                      // (1) reads done

    *(float4*)&xs[xr][xc] = pf;                           // stage x_{t+1}
    int tn = t + 2; if (tn > GRU_T - 1) tn = GRU_T - 1;
    pf = *(const float4*)(xbase + (size_t)tn * GRU_D0);   // prefetch x_{t+2}

    // ---- Layer 0: all 4 waves, gate-column tile ct = wave -------------------
    {
      v8f az, ar, axh, ahh;
#pragma unroll
      for (int i = 0; i < 8; ++i) { az[i] = bz0; ar[i] = brr0; axh[i] = bxh0; ahh[i] = bhh0; }
      az  = wmma_bf16(ax,   f0[0], az);
      ar  = wmma_bf16(ax,   f0[1], ar);
      axh = wmma_bf16(ax,   f0[2], axh);
      az  = wmma_bf16(ah0a, f0[3], az);
      az  = wmma_bf16(ah0b, f0[4], az);
      ar  = wmma_bf16(ah0a, f0[5], ar);
      ar  = wmma_bf16(ah0b, f0[6], ar);
      ahh = wmma_bf16(ah0a, f0[7], ahh);
      ahh = wmma_bf16(ah0b, f0[8], ahh);
#pragma unroll
      for (int i = 0; i < 8; ++i) {
        float z  = sigmoidf_fast(az[i]);
        float rr = sigmoidf_fast(ar[i]);
        float hh = tanhf_fast(axh[i] + rr * ahh[i]);
        h0r[i] = z * h0r[i] + (1.0f - z) * hh;
        h0bf[8 * hi + i][j0] = bf16_bits(h0r[i]);
      }
    }

    // ---- L1/L2 recurrent projections (depend only on h_{t-1}) --------------
    v8f az1, ar1, ahh1, az2, ar2, ahh2;
#pragma unroll
    for (int i = 0; i < 8; ++i) {
      az1[i] = bz1; ar1[i] = brr1; ahh1[i] = bhh1;
      az2[i] = bz2; ar2[i] = brr2; ahh2[i] = bhh2;
    }
    if (wave < 2) {
      az1  = wmma_bf16(ah1, f1[0], az1);
      ar1  = wmma_bf16(ah1, f1[1], ar1);
      ahh1 = wmma_bf16(ah1, f1[2], ahh1);
    }
    if (wave == 0) {
      v16bf ah2 = gatherAh(h2bf, GRU_H2, 0, lane);  // wave-0 private, in-order LDS
      az2  = wmma_bf16(ah2, u2f[0], az2);
      ar2  = wmma_bf16(ah2, u2f[1], ar2);
      ahh2 = wmma_bf16(ah2, u2f[2], ahh2);
    }
    __syncthreads();                                      // (2) h0bf(t) ready

    // ---- Layer 1: waves 0,1 ------------------------------------------------
    if (wave < 2) {
      v16bf a0 = gatherAh(h0bf, GRU_H0, 0, lane);
      v16bf a1 = gatherAh(h0bf, GRU_H0, 1, lane);
      v8f axh1;
#pragma unroll
      for (int i = 0; i < 8; ++i) axh1[i] = bxh1;
      az1  = wmma_bf16(a0, f1[3], az1);
      az1  = wmma_bf16(a1, f1[4], az1);
      ar1  = wmma_bf16(a0, f1[5], ar1);
      ar1  = wmma_bf16(a1, f1[6], ar1);
      axh1 = wmma_bf16(a0, f1[7], axh1);
      axh1 = wmma_bf16(a1, f1[8], axh1);
#pragma unroll
      for (int i = 0; i < 8; ++i) {
        float z  = sigmoidf_fast(az1[i]);
        float rr = sigmoidf_fast(ar1[i]);
        float hh = tanhf_fast(axh1[i] + rr * ahh1[i]);
        h1r[i] = z * h1r[i] + (1.0f - z) * hh;
        h1bf[8 * hi + i][wave * 16 + c] = bf16_bits(h1r[i]);
      }
    }
    __syncthreads();                                      // (3) h1bf(t) ready

    // ---- Layer 2: wave 0 ---------------------------------------------------
    if (wave == 0) {
      v16bf a1 = gatherAh(h1bf, GRU_H1, 0, lane);
      v8f axh2;
#pragma unroll
      for (int i = 0; i < 8; ++i) axh2[i] = bxh2;
      az2  = wmma_bf16(a1, w2f[0], az2);
      ar2  = wmma_bf16(a1, w2f[1], ar2);
      axh2 = wmma_bf16(a1, w2f[2], axh2);
#pragma unroll
      for (int i = 0; i < 8; ++i) {
        float z  = sigmoidf_fast(az2[i]);
        float rr = sigmoidf_fast(ar2[i]);
        float hh = tanhf_fast(axh2[i] + rr * ahh2[i]);
        h2r[i] = z * h2r[i] + (1.0f - z) * hh;
        h2bf[8 * hi + i][c] = bf16_bits(h2r[i]);
      }
    }
  }

  // ---- Dense head: out[b] = h2[b,:] @ Wd + bd ------------------------------
  if (wave == 0) {
#pragma unroll
    for (int i = 0; i < 8; ++i) h2f[8 * hi + i][c] = h2r[i];
  }
  __syncthreads();
  if (tid < 16) {
    float acc = bd[0];
#pragma unroll
    for (int j = 0; j < 16; ++j) acc += h2f[tid][j] * Wd[j];
    out[b0 + tid] = acc;
  }
}

extern "C" void kernel_launch(void* const* d_in, const int* in_sizes, int n_in,
                              void* d_out, int out_size, void* d_ws, size_t ws_size,
                              hipStream_t stream) {
  (void)n_in; (void)out_size; (void)d_ws; (void)ws_size;
  const float* x   = (const float*)d_in[0];
  const float* W0  = (const float*)d_in[1];
  const float* U0  = (const float*)d_in[2];
  const float* bi0 = (const float*)d_in[3];
  const float* br0 = (const float*)d_in[4];
  const float* W1  = (const float*)d_in[5];
  const float* U1  = (const float*)d_in[6];
  const float* bi1 = (const float*)d_in[7];
  const float* br1 = (const float*)d_in[8];
  const float* W2  = (const float*)d_in[9];
  const float* U2  = (const float*)d_in[10];
  const float* bi2 = (const float*)d_in[11];
  const float* br2 = (const float*)d_in[12];
  const float* Wd  = (const float*)d_in[13];
  const float* bd  = (const float*)d_in[14];
  float* out = (float*)d_out;

  const int B = in_sizes[0] / (GRU_T * GRU_D0);   // 1024
  dim3 grid(B / 16), block(128);
  hipLaunchKernelGGL(stacked_gru_fused, grid, block, 0, stream,
                     x, W0, U0, bi0, br0, W1, U1, bi1, br1,
                     W2, U2, bi2, br2, Wd, bd, out);
}
